// LigEdgeBuilder_31001073943182
// MI455X (gfx1250) — matrix operand order repaired and checked by reading, not compiled, not run on hardware
//
#include <hip/hip_runtime.h>

// ---------------------------------------------------------------------------
// LigEdgeBuilder on gfx1250 (MI455X):
//   edge_vec -> |d| -> GaussianSmearing(50) -> [E,50]@W1[50,64] -> relu
//   -> [E,64]@W2[64,64] -> + lig_ea   (output [E,64] f32, plus edge_index)
//
// Main kernel: v_wmma_f32_16x16x32_f16, one wave per FULL 16-edge tile,
// single straight-line path (no tail clamps/guards at all). W1/W2 staged
// once per workgroup into LDS in per-lane B-operand layout; h transposed
// C->A layout through padded LDS. Leftover edges (E % 16) go to a tiny
// rolled scalar tail kernel (zero blocks when E is a multiple of 16).
// ---------------------------------------------------------------------------

typedef __attribute__((ext_vector_type(16))) _Float16 v16h;
typedef __attribute__((ext_vector_type(8)))  _Float16 v8h;
typedef __attribute__((ext_vector_type(8)))  float    v8f;

#define RAD      50          // RADIUS_EMB_DIM
#define FD       64          // FOLD_DIM
#define HSTRIDE  72          // padded h-tile row stride in halfs (144B)
#define WSLOT    24          // halfs per (operand,lane) weight slot (48B)
#define NOPER    16          // 2 weight mats * 4 N-tiles * 2 K-blocks
#define WAVES_PER_BLOCK 8

__global__ __launch_bounds__(256) void lig_edge_wmma_kernel(
    const float* __restrict__ pos,
    const float* __restrict__ lig_ea,
    const int*   __restrict__ edge_index,
    const float* __restrict__ W1,
    const float* __restrict__ b1,
    const float* __restrict__ W2,
    const float* __restrict__ b2,
    float* __restrict__ out_attr,
    int E,                // for edge_index second-row offset
    int fullTiles)        // number of complete 16-edge tiles
{
    const float STEP  = 30.0f / 49.0f;              // linspace(0,30,50) spacing
    const float COEFF = -0.5f / (STEP * STEP);      // gaussian coefficient

    const int lane = threadIdx.x & 31;
    const int wid  = threadIdx.x >> 5;
    const int half = lane >> 4;                      // 0: lanes 0-15, 1: lanes 16-31
    const int m    = lane & 15;                      // row within 16x16 tile

    // Weight operands in B layout (24 KB) + per-wave h tiles (18 KB) = 42 KB.
    __shared__ _Float16 wlds[NOPER * 32 * WSLOT];
    __shared__ _Float16 hbuf[WAVES_PER_BLOCK * 16 * HSTRIDE];
    _Float16* hb = hbuf + wid * 16 * HSTRIDE;

    // ---- Cooperative staging of W1/W2 into per-lane B-operand layout ------
    // 16-bit B 32x16: element t of the v16h operand for (ntile, kblk) holds
    // W[kblk*32 + half*8 + t (+8 if t>=8)][ntile*16 + (lane&15)].
    for (int ei = threadIdx.x; ei < NOPER * 32 * 16; ei += blockDim.x) {
        const int t     = ei & 15;
        const int ln    = (ei >> 4) & 31;
        const int kb    = (ei >> 9) & 1;
        const int nt    = (ei >> 10) & 3;
        const int which = (ei >> 12) & 1;
        const int hf    = ln >> 4;
        const int mm    = ln & 15;
        const int n     = nt * 16 + mm;
        const int k     = kb * 32 + hf * 8 + t + ((t >= 8) ? 8 : 0);
        float v;
        if (which) v = W2[k * FD + n];
        else       v = (k < RAD) ? W1[k * FD + n] : 0.0f;   // pad K 50..63
        const int o = which * 8 + nt * 2 + kb;
        wlds[(o * 32 + ln) * WSLOT + t] = (_Float16)v;
    }
    __syncthreads();

    // per-lane bias broadcast for each N tile
    float c1[4], c2[4];
#pragma unroll
    for (int nt = 0; nt < 4; ++nt) {
        c1[nt] = b1[nt * 16 + m];
        c2[nt] = b2[nt * 16 + m];
    }

    // per-lane element offsets for the C-layout rows of a tile (tile-invariant)
    int off[8];
#pragma unroll
    for (int r = 0; r < 8; ++r) off[r] = (r + half * 8) * FD + m;

    // fetch one register-resident B operand from the LDS staging area
    auto loadW = [&](int o) -> v16h {
        const _Float16* p = wlds + (o * 32 + lane) * WSLOT;
        const v8h lo = *(const v8h*)(p);
        const v8h hi = *(const v8h*)(p + 8);
        return __builtin_shufflevector(lo, hi, 0,1,2,3,4,5,6,7,8,9,10,11,12,13,14,15);
    };

    const int wavesTotal = gridDim.x * (blockDim.x >> 5);

    for (int tile = blockIdx.x * (blockDim.x >> 5) + wid; tile < fullTiles; tile += wavesTotal) {
        // prefetch next tile's lig_ea block into the cache hierarchy
        if (tile + wavesTotal < fullTiles) {
            __builtin_prefetch(lig_ea + (long long)(tile + wavesTotal) * 16 * FD + lane * 16, 0, 1);
        }

        // ---- distance for this lane's row (lane and lane+16 share a row) --
        const int e  = tile * 16 + m;                // always < E (full tiles only)
        const int s  = edge_index[e];
        const int dd = edge_index[E + e];
        const float dx = pos[dd * 3 + 0] - pos[s * 3 + 0];
        const float dy = pos[dd * 3 + 1] - pos[s * 3 + 1];
        const float dz = pos[dd * 3 + 2] - pos[s * 3 + 2];
        const float dist = sqrtf(dx * dx + dy * dy + dz * dz);

        // ---- Gaussian basis directly in 16-bit A-operand layout -----------
        // A 16x32: element t -> K = kblk*32 + half*8 + t (+8 if t>=8)
        v16h a0, a1;
#pragma unroll
        for (int t = 0; t < 16; ++t) {
            const int k0 = half * 8 + t + ((t >= 8) ? 8 : 0);
            const float df0 = dist - (float)k0 * STEP;
            a0[t] = (_Float16)__expf(COEFF * df0 * df0);
            const int k1 = 32 + k0;
            const float df1 = dist - (float)k1 * STEP;
            const float g1 = (k1 < RAD) ? __expf(COEFF * df1 * df1) : 0.0f;
            a1[t] = (_Float16)g1;
        }

        // ---- GEMM1: h = relu(g @ W1 + b1) ; spill f16 tile to LDS ---------
#pragma unroll
        for (int nt = 0; nt < 4; ++nt) {
            const v16h w0 = loadW(nt * 2 + 0);
            const v16h w1 = loadW(nt * 2 + 1);
            v8f c;
#pragma unroll
            for (int r = 0; r < 8; ++r) c[r] = c1[nt];
            c = __builtin_amdgcn_wmma_f32_16x16x32_f16(false, a0, false, w0, (short)0, c, false, false);
            c = __builtin_amdgcn_wmma_f32_16x16x32_f16(false, a1, false, w1, (short)0, c, false, false);
#pragma unroll
            for (int r = 0; r < 8; ++r) {
                // relu in one VALU op: med3(x, 0, +inf) clamps to [0, inf)
                const float hv = __builtin_amdgcn_fmed3f(c[r], 0.0f, __builtin_inff());
                // C layout: VGPR r, lane -> (M = r + half*8, N = nt*16 + m)
                hb[(r + half * 8) * HSTRIDE + nt * 16 + m] = (_Float16)hv;
            }
        }

        // ---- reload h in A-operand layout (LDS transpose, intra-wave) -----
        const v8h lo0 = *(const v8h*)(hb + m * HSTRIDE +      half * 8);
        const v8h hi0 = *(const v8h*)(hb + m * HSTRIDE +      half * 8 + 16);
        const v8h lo1 = *(const v8h*)(hb + m * HSTRIDE + 32 + half * 8);
        const v8h hi1 = *(const v8h*)(hb + m * HSTRIDE + 32 + half * 8 + 16);
        const v16h ha0 = __builtin_shufflevector(lo0, hi0, 0,1,2,3,4,5,6,7,8,9,10,11,12,13,14,15);
        const v16h ha1 = __builtin_shufflevector(lo1, hi1, 0,1,2,3,4,5,6,7,8,9,10,11,12,13,14,15);

        // ---- GEMM2: emb = h @ W2 + (b2 + lig_ea) ; store ------------------
        const float* lp = lig_ea   + (long long)tile * 16 * FD;
        float*       op = out_attr + (long long)tile * 16 * FD;

        v8f cin[4];
#pragma unroll
        for (int nt = 0; nt < 4; ++nt)
#pragma unroll
            for (int r = 0; r < 8; ++r)
                cin[nt][r] = lp[off[r] + nt * 16] + c2[nt];   // saddr+voff+imm

#pragma unroll
        for (int nt = 0; nt < 4; ++nt) {
            const v16h w0 = loadW(8 + nt * 2 + 0);
            const v16h w1 = loadW(8 + nt * 2 + 1);
            v8f c = cin[nt];
            c = __builtin_amdgcn_wmma_f32_16x16x32_f16(false, ha0, false, w0, (short)0, c, false, false);
            c = __builtin_amdgcn_wmma_f32_16x16x32_f16(false, ha1, false, w1, (short)0, c, false, false);
#pragma unroll
            for (int r = 0; r < 8; ++r) op[off[r] + nt * 16] = c[r];
        }
    }
}

// ---------------------------------------------------------------------------
// Rolled scalar tail kernel for the E%16 leftover edges (usually 0 blocks).
// One thread per (edge, out-column).
// ---------------------------------------------------------------------------
__global__ void lig_edge_tail_kernel(
    const float* __restrict__ pos, const float* __restrict__ lig_ea,
    const int* __restrict__ edge_index,
    const float* __restrict__ W1, const float* __restrict__ b1,
    const float* __restrict__ W2, const float* __restrict__ b2,
    float* __restrict__ out_attr, int E, int start)
{
    const float STEP  = 30.0f / 49.0f;
    const float COEFF = -0.5f / (STEP * STEP);
    const int work = (E - start) * FD;
    const int i = blockIdx.x * blockDim.x + threadIdx.x;
    if (i >= work) return;
    const int e = start + i / FD;
    const int n = i % FD;
    const int s = edge_index[e], dd = edge_index[E + e];
    const float dx = pos[dd * 3 + 0] - pos[s * 3 + 0];
    const float dy = pos[dd * 3 + 1] - pos[s * 3 + 1];
    const float dz = pos[dd * 3 + 2] - pos[s * 3 + 2];
    const float dist = sqrtf(dx * dx + dy * dy + dz * dz);
    float g[RAD];
#pragma unroll 1
    for (int k = 0; k < RAD; ++k) {
        const float df = dist - (float)k * STEP;
        g[k] = __expf(COEFF * df * df);
    }
    float acc = b2[n];
#pragma unroll 1
    for (int j = 0; j < FD; ++j) {
        float h = b1[j];
#pragma unroll 1
        for (int k = 0; k < RAD; ++k) h += g[k] * W1[k * FD + j];
        h = fmaxf(h, 0.0f);
        acc += h * W2[j * FD + n];
    }
    out_attr[e * FD + n] = lig_ea[e * FD + n] + acc;
}

// Pass-through edge_index widened to int64 slots if the harness reserved 8B/elt.
__global__ void widen_idx_kernel(const int* __restrict__ idx,
                                 long long* __restrict__ out, int n)
{
    const int i = blockIdx.x * blockDim.x + threadIdx.x;
    if (i < n) out[i] = (long long)idx[i];
}

extern "C" void kernel_launch(void* const* d_in, const int* in_sizes, int n_in,
                              void* d_out, int out_size, void* d_ws, size_t ws_size,
                              hipStream_t stream)
{
    const float* pos    = (const float*)d_in[0];
    const float* lig_ea = (const float*)d_in[1];
    const int*   eidx   = (const int*)  d_in[2];
    const float* W1     = (const float*)d_in[3];
    const float* b1     = (const float*)d_in[4];
    const float* W2     = (const float*)d_in[5];
    const float* b2     = (const float*)d_in[6];

    const int E = in_sizes[2] / 2;                       // edge_index is [2, E]
    const long long attrElems = (long long)E * FD;
    const long long extra     = (long long)out_size - attrElems;

    float* out  = (float*)d_out;
    float* attr = out;
    if (extra == (long long)2 * E) {
        // edge_index stored as 4B slots: raw device-to-device copy
        hipMemcpyAsync(d_out, d_in[2], sizeof(int) * (size_t)2 * E,
                       hipMemcpyDeviceToDevice, stream);
        attr = out + (long long)2 * E;
    } else if (extra == (long long)4 * E) {
        // edge_index stored as 8B (int64) slots: widen
        const int n = 2 * E;
        widen_idx_kernel<<<(n + 255) / 256, 256, 0, stream>>>(eidx, (long long*)d_out, n);
        attr = out + (long long)4 * E;
    } else if (extra > 0) {
        attr = out + extra;                              // unknown header layout: skip it
    }

    const int fullTiles = E / 16;
    const int rem       = E - fullTiles * 16;

    if (fullTiles > 0) {
        int blocks = (fullTiles + WAVES_PER_BLOCK - 1) / WAVES_PER_BLOCK;
        if (blocks > 4096) blocks = 4096;                // grid-stride over tiles
        lig_edge_wmma_kernel<<<blocks, WAVES_PER_BLOCK * 32, 0, stream>>>(
            pos, lig_ea, eidx, W1, b1, W2, b2, attr, E, fullTiles);
    }
    if (rem > 0) {
        const int work = rem * FD;
        lig_edge_tail_kernel<<<(work + 255) / 256, 256, 0, stream>>>(
            pos, lig_ea, eidx, W1, b1, W2, b2, attr, E, fullTiles * 16);
    }
}
